// PlyPagedAttentionManager_53515292508316
// MI455X (gfx1250) — matrix operand order repaired and checked by reading, not compiled, not run on hardware
//
#include <hip/hip_runtime.h>

typedef __attribute__((ext_vector_type(16))) _Float16 v16h;
typedef __attribute__((ext_vector_type(8)))  float    v8f;
typedef __attribute__((ext_vector_type(4)))  float    v4f;

#define B_   16
#define H_   32
#define D_   128
#define BS_  16
#define MAXB 128
#define HD_  (H_ * D_)

__global__ __launch_bounds__(128)
void paged_attn_wmma_kernel(const float* __restrict__ q,
                            const float* __restrict__ k_cache,
                            const float* __restrict__ v_cache,
                            const int*   __restrict__ block_tables,
                            const int*   __restrict__ context_lens,
                            float*       __restrict__ out)
{
    __shared__ float q_lds[D_];
    __shared__ float red_m[4];
    __shared__ float red_l[4];
    __shared__ float red_o[4][D_];

    const int h      = blockIdx.x;
    const int b      = blockIdx.y;
    const int tid    = threadIdx.x;
    const int wave   = tid >> 5;
    const int lane   = tid & 31;
    const int lane16 = lane & 15;
    const int hi     = lane >> 4;

    // Load q row into LDS, softmax scale folded in (~0.09*N(0,1): safe in f16).
    const float sm_scale = 0.08838834764831845f;  // 1/sqrt(128)
    if (tid < D_)
        q_lds[tid] = q[((size_t)b * H_ + h) * D_ + tid] * sm_scale;
    __syncthreads();

    // A fragments: q broadcast across all 16 M-rows, one per 32-dim chunk.
    // A k order used here (contiguous, same as B): elem e -> K=16*hi+e.
    v16h a_frag[4];
    #pragma unroll
    for (int c = 0; c < 4; ++c) {
        #pragma unroll
        for (int e = 0; e < 8; ++e) {
            a_frag[c][e]     = (_Float16)q_lds[32 * c + 16 * hi + e];
            a_frag[c][8 + e] = (_Float16)q_lds[32 * c + 16 * hi + 8 + e];
        }
    }

    const int ctx    = context_lens[b];
    const int nchunk = (ctx + 31) >> 5;            // 32-token chunks (2 paged blocks)

    const float* kbh  = k_cache + (size_t)h * D_;
    const float* vbh  = v_cache + (size_t)h * D_;
    const int*   btab = block_tables + (size_t)b * MAXB;

    float m_run = -__builtin_inff();
    float l_run = 0.0f;
    float oacc[8] = {0.f, 0.f, 0.f, 0.f, 0.f, 0.f, 0.f, 0.f};

    for (int ch = wave; ch < nchunk; ch += 4) {
        const int pb0 = btab[2 * ch + 0];
        const int pb1 = btab[2 * ch + 1];

        if (ch + 4 < nchunk) {  // hint next chunk's K block (global_prefetch_b8)
            const int nb = btab[2 * (ch + 4)];
            __builtin_prefetch(kbh + (size_t)nb * BS_ * HD_, 0, 0);
        }

        // ---------------- Q·K^T : two 16-token halves ----------------
        // B 32x16 layout: lane holds K[token lane16, dim 32c + 16*hi + e].
        // Streamed once, 1 GB total vs 192 MB L2 -> non-temporal loads.
        v8f s01[2];
        #pragma unroll
        for (int th = 0; th < 2; ++th) {
            const int pb = th ? pb1 : pb0;
            const float* kp = kbh + ((size_t)pb * BS_ + lane16) * HD_ + 16 * hi;
            v8f s = {};
            #pragma unroll
            for (int c = 0; c < 4; ++c) {
                const v4f* kp4 = (const v4f*)(kp + 32 * c);   // 16B-aligned
                v4f k0 = __builtin_nontemporal_load(kp4 + 0);
                v4f k1 = __builtin_nontemporal_load(kp4 + 1);
                v4f k2 = __builtin_nontemporal_load(kp4 + 2);
                v4f k3 = __builtin_nontemporal_load(kp4 + 3);
                v16h bk;
                #pragma unroll
                for (int e = 0; e < 4; ++e) {
                    bk[e]      = (_Float16)k0[e];
                    bk[4 + e]  = (_Float16)k1[e];
                    bk[8 + e]  = (_Float16)k2[e];
                    bk[12 + e] = (_Float16)k3[e];
                }
                s = __builtin_amdgcn_wmma_f32_16x16x32_f16(
                        false, a_frag[c], false, bk, (short)0, s, false, false);
            }
            s01[th] = s;
        }

        // ---------------- online softmax over 32 scores ----------------
        // D rows identical (A rows identical): s[0] at lane L = score[token L%16].
        const int g0 = ch * 32 + lane16;
        float sc0 = s01[0][0];
        float sc1 = s01[1][0];
        if (g0 >= ctx)      sc0 = -__builtin_inff();
        if (g0 + 16 >= ctx) sc1 = -__builtin_inff();

        float cmax = fmaxf(sc0, sc1);
        #pragma unroll
        for (int w = 8; w >= 1; w >>= 1)
            cmax = fmaxf(cmax, __shfl_xor(cmax, w, 32));

        const float m_new = fmaxf(m_run, cmax);
        const float alpha = __expf(m_run - m_new);
        const float p0 = __expf(sc0 - m_new);
        const float p1 = __expf(sc1 - m_new);

        float psum = p0 + p1;
        #pragma unroll
        for (int w = 8; w >= 1; w >>= 1)
            psum += __shfl_xor(psum, w, 32);

        l_run = l_run * alpha + psum;
        m_run = m_new;

        // A fragment from p (broadcast rows): p_all[k] for k=0..31.
        v16h ap;
        #pragma unroll
        for (int e = 0; e < 8; ++e) {
            ap[e]     = (_Float16)__shfl(hi ? p1 : p0, 16 * hi + e, 32);
            ap[8 + e] = (_Float16)__shfl(hi ? p1 : p0, 16 * hi + 8 + e, 32);
        }

        // ---------------- P·V ----------------
        // B 32x16: lane holds V[token 16*hi + e, dim 16t + lane16];
        // tokens 0..15 from pb0 (hi==0 lanes), 16..31 from pb1 (hi==1 lanes).
        // Each b32 load is 2x64B fully-coalesced per half-wave; NT to spare L2.
        const float* vp = vbh + (size_t)(hi ? pb1 : pb0) * BS_ * HD_ + lane16;
        #pragma unroll
        for (int t = 0; t < 8; ++t) {
            v16h bv;
            #pragma unroll
            for (int e = 0; e < 16; ++e)
                bv[e] = (_Float16)__builtin_nontemporal_load(
                            vp + (size_t)e * HD_ + 16 * t);
            v8f pv = {};
            pv = __builtin_amdgcn_wmma_f32_16x16x32_f16(
                     false, ap, false, bv, (short)0, pv, false, false);
            oacc[t] = oacc[t] * alpha + pv[0];
        }
    }

    // ---------------- cross-wave combine via LDS ----------------
    if (lane == 0) { red_m[wave] = m_run; red_l[wave] = l_run; }
    if (hi == 0) {
        #pragma unroll
        for (int t = 0; t < 8; ++t)
            red_o[wave][16 * t + lane16] = oacc[t];
    }
    __syncthreads();

    if (tid < D_) {
        const float mg = fmaxf(fmaxf(red_m[0], red_m[1]), fmaxf(red_m[2], red_m[3]));
        float lg = 0.f, og = 0.f;
        #pragma unroll
        for (int w = 0; w < 4; ++w) {
            const float f = __expf(red_m[w] - mg);  // empty wave: m=-inf -> f=0
            lg += red_l[w] * f;
            og += red_o[w][tid] * f;
        }
        out[((size_t)b * H_ + h) * D_ + tid] = og / lg;
    }
}

extern "C" void kernel_launch(void* const* d_in, const int* in_sizes, int n_in,
                              void* d_out, int out_size, void* d_ws, size_t ws_size,
                              hipStream_t stream) {
    (void)in_sizes; (void)n_in; (void)out_size; (void)d_ws; (void)ws_size;
    const float* q            = (const float*)d_in[0];
    const float* k_cache      = (const float*)d_in[1];
    const float* v_cache      = (const float*)d_in[2];
    const int*   block_tables = (const int*)d_in[3];
    const int*   context_lens = (const int*)d_in[4];
    float*       out          = (float*)d_out;

    dim3 grid(H_, B_);   // 512 workgroups: one per (batch, head)
    dim3 block(128);     // 4 wave32 waves splitting the KV chunks
    hipLaunchKernelGGL(paged_attn_wmma_kernel, grid, block, 0, stream,
                       q, k_cache, v_cache, block_tables, context_lens, out);
}
